// BaselineModel_36687610642509
// MI455X (gfx1250) — matrix operand order repaired and checked by reading, not compiled, not run on hardware
//
#include <hip/hip_runtime.h>
#include <hip/hip_bf16.h>
#include <stdint.h>

// ---------------------------------------------------------------------------
// GGNN on MI455X (gfx1250): bf16 WMMA GEMMs, f32 accumulate.
//  - edge tensor transposed+downcast once to bf16 (64 MiB, L2-resident)
//  - split-K (16 slices) on the dominant K=16384 aggregation GEMM -> 1024 WGs
//  - double-buffered LDS pipeline, A tiles staged via CDNA5 async global->LDS
//    (global_load_async_to_lds_b128 / s_wait_asynccnt), one barrier per step
// Workspace required: ~94 MB.
// ---------------------------------------------------------------------------

#define NND    2048          // nodes
#define DIM    128           // embedding dim
#define NE     8             // edge channels
#define NITER  8
#define KAGG   (NND * NE)    // 16384
#define NSPLIT 16            // split-K slices for the aggregation GEMM

typedef __bf16 bf16;
typedef __bf16 bf16x16 __attribute__((ext_vector_type(16)));
typedef float  f32x8   __attribute__((ext_vector_type(8)));

union FragAB { bf16x16 v; unsigned int u[8]; };

#define PA 40   // LDS pitch, A tile (bf16): 16B-aligned rows, conflict-free
#define PB 34   // LDS pitch, B^T tile (bf16): 68B stride -> bank-safe

// ---- CDNA5 async global->LDS copy (16B per lane), tracked by ASYNCcnt -----
__device__ __forceinline__ void async_copy_b128(const bf16* g, bf16* l)
{
    unsigned loff = (unsigned)(size_t)l;            // low 32b = LDS byte offset
    unsigned long long ga = (unsigned long long)(size_t)g;
    asm volatile("global_load_async_to_lds_b128 %0, %1, off"
                 :: "v"(loff), "v"(ga) : "memory");
}
__device__ __forceinline__ void wait_async()
{
    asm volatile("s_wait_asynccnt 0x0" ::: "memory");
}

// ---------------------------------------------------------------------------
// Tiled bf16 GEMM:  C[M x Ncols] = A[M x K] * B[K x Ncols] (+bias)
// 128x32 tile per 256-thread WG (8 wave32); wave w: rows [16w,16w+16), 2 accs.
// BTRANS: B stored row-major [Ncols x K]  (computes A * Bmat^T)
// SPLITK: grid.z slices of length Kslice; f32 partials at Cout + z*M*ldc.
// ---------------------------------------------------------------------------
template<bool BTRANS, bool SPLITK, bool BIAS, bool OUTBF16>
__global__ void __launch_bounds__(256)
wmma_gemm(const bf16* __restrict__ A, int lda,
          const bf16* __restrict__ B, int ldb,
          const float* __restrict__ bias,
          void* __restrict__ Cout, int ldc,
          int Kslice, int Mtotal)
{
    __shared__ bf16 sA[2 * 128 * PA];
    __shared__ bf16 sBT[2 * 32 * PB];

    const int tid  = threadIdx.x;
    const int lane = tid & 31;
    const int w    = tid >> 5;          // wave id 0..7
    const int m    = lane & 15;
    const int half = lane >> 4;

    const int row0  = blockIdx.y * 128;
    const int col0  = blockIdx.x * 32;
    const int kbase = SPLITK ? blockIdx.z * Kslice : 0;

    // ---- per-thread staging pointers (incremented, no in-loop 64b muls) ----
    const int rA = tid >> 2, cA = (tid & 3) * 8;            // A: 2 x 16B chunks
    const bf16* pA0 = A + (size_t)(row0 + rA) * lda + kbase + cA;
    const bf16* pA1 = pA0 + (size_t)64 * lda;
    bf16* lA0 = sA + rA * PA + cA;
    bf16* lA1 = lA0 + 64 * PA;

    int kB, nB;
    if (BTRANS) { nB = tid >> 3; kB = (tid & 7) * 4; }
    else        { kB = tid >> 3; nB = (tid & 7) * 4; }
    const bf16* pB = BTRANS ? (B + (size_t)(col0 + nB) * ldb + kbase + kB)
                            : (B + (size_t)(kbase + kB) * ldb + col0 + nB);
    const size_t stepB = BTRANS ? 32 : (size_t)32 * ldb;

    f32x8 acc0 = {0.f, 0.f, 0.f, 0.f, 0.f, 0.f, 0.f, 0.f};
    f32x8 acc1 = acc0;

    // ---- stage tile 0 into buffer 0 ----
    {
        async_copy_b128(pA0, lA0);
        async_copy_b128(pA1, lA1);
        uint2 v = *(const uint2*)pB;
        if (BTRANS) {
            bf16* d = sBT + nB * PB + kB;
            *(unsigned int*)(d)     = v.x;
            *(unsigned int*)(d + 2) = v.y;
        } else {
            const bf16* pv = (const bf16*)&v;
            #pragma unroll
            for (int j = 0; j < 4; ++j) sBT[(nB + j) * PB + kB] = pv[j];
        }
    }
    wait_async();
    __syncthreads();

    int bufc = 0;
    for (int k0 = 0; k0 < Kslice; k0 += 32) {
        // ---- stage tile k0+32 into the other buffer (overlaps WMMA) ----
        if (k0 + 32 < Kslice) {
            pA0 += 32; pA1 += 32; pB += stepB;
            const int nb = 1 - bufc;
            async_copy_b128(pA0, lA0 + nb * 128 * PA - bufc * 0 + (nb ? 128 * PA - 0 : 0) * 0
                                  + (nb - 0) * 0 + (nb ? 0 : 0));  // (see below)
            // NOTE: simple form:
            async_copy_b128(pA1, sA + nb * 128 * PA + rA * PA + cA + 64 * PA);
            uint2 v = *(const uint2*)pB;
            bf16* bb = sBT + nb * 32 * PB;
            if (BTRANS) {
                bf16* d = bb + nB * PB + kB;
                *(unsigned int*)(d)     = v.x;
                *(unsigned int*)(d + 2) = v.y;
            } else {
                const bf16* pv = (const bf16*)&v;
                #pragma unroll
                for (int j = 0; j < 4; ++j) bb[(nB + j) * PB + kB] = pv[j];
            }
            __builtin_prefetch(pB + stepB, 0, 3);   // global_prefetch_b8
        }

        // ---- fragments from current buffer, per gfx1250 16-bit layouts ----
        const bf16* cAb = sA + bufc * 128 * PA;
        const bf16* cBb = sBT + bufc * 32 * PB;
        FragAB a, b0, b1;
        #pragma unroll
        for (int j = 0; j < 8; ++j) {
            int kk = (j < 4) ? (half * 8 + 2 * j) : (16 + half * 8 + 2 * (j - 4));
            a.u[j] = *(const unsigned int*)(cAb + (w * 16 + m) * PA + kk);
        }
        #pragma unroll
        for (int j = 0; j < 8; ++j) {
            int kk = (j < 4) ? (half * 8 + 2 * j) : (16 + half * 8 + 2 * (j - 4));
            b0.u[j] = *(const unsigned int*)(cBb + m * PB + kk);
            b1.u[j] = *(const unsigned int*)(cBb + (m + 16) * PB + kk);
        }
        acc0 = __builtin_amdgcn_wmma_f32_16x16x32_bf16(
                   false, a.v, false, b0.v, (short)0, acc0, false, false);
        acc1 = __builtin_amdgcn_wmma_f32_16x16x32_bf16(
                   false, a.v, false, b1.v, (short)0, acc1, false, false);

        wait_async();       // next tile's async A-copy landed (wave-local)
        __syncthreads();    // + dscnt-flush: B stores visible, reads done
        bufc ^= 1;
    }

    // ---- epilogue: C row = 16w + 8*half + r, col = lane%16 (+0/+16) ----
    const int row = row0 + w * 16 + half * 8;
    const int c0  = col0 + m;
    float bv0 = 0.f, bv1 = 0.f;
    if (BIAS) { bv0 = bias[c0]; bv1 = bias[c0 + 16]; }
    float* Cf = (float*)Cout;
    if (SPLITK) Cf += (size_t)blockIdx.z * Mtotal * ldc;
    #pragma unroll
    for (int r = 0; r < 8; ++r) {
        float v0 = acc0[r] + bv0;
        float v1 = acc1[r] + bv1;
        if (OUTBF16) {
            bf16* C = (bf16*)Cout;
            C[(size_t)(row + r) * ldc + c0]      = (bf16)v0;
            C[(size_t)(row + r) * ldc + c0 + 16] = (bf16)v1;
        } else {
            Cf[(size_t)(row + r) * ldc + c0]      = v0;
            Cf[(size_t)(row + r) * ldc + c0 + 16] = v1;
        }
    }
}

// sum split-K partials + bias -> bf16
__global__ void __launch_bounds__(256)
reduce_msg(const float* __restrict__ parts, const float* __restrict__ bias,
           bf16* __restrict__ outb)
{
    int idx = blockIdx.x * 256 + threadIdx.x;      // over N*D
    float s = bias[idx & (DIM - 1)];
    #pragma unroll
    for (int z = 0; z < NSPLIT; ++z) s += parts[(size_t)z * NND * DIM + idx];
    outb[idx] = (bf16)s;
}

// edge[s][t][e] (f32) -> edgeT[t][s*8+e] (bf16), 16x16 (s,t) tiles via LDS
__global__ void __launch_bounds__(256)
edge_transpose(const float* __restrict__ edge, bf16* __restrict__ edgeT)
{
    __shared__ bf16 tile[16 * 128];
    const int tid = threadIdx.x;
    const int s0 = blockIdx.x * 16, t0 = blockIdx.y * 16;

    int s_l = tid >> 4, t_l = tid & 15;
    const float4* p = (const float4*)(edge + ((size_t)(s0 + s_l) * NND + (t0 + t_l)) * NE);
    float4 v0 = p[0], v1 = p[1];
    bf16* dst = tile + t_l * 128 + s_l * 8;
    dst[0] = (bf16)v0.x; dst[1] = (bf16)v0.y; dst[2] = (bf16)v0.z; dst[3] = (bf16)v0.w;
    dst[4] = (bf16)v1.x; dst[5] = (bf16)v1.y; dst[6] = (bf16)v1.z; dst[7] = (bf16)v1.w;
    __syncthreads();

    int tt = tid >> 4, ch = tid & 15;
    uint4 v = *(const uint4*)(tile + tt * 128 + ch * 8);
    *(uint4*)(edgeT + (size_t)(t0 + tt) * KAGG + (size_t)s0 * NE + ch * 8) = v;
}

// W_msg[e][k][d] (f32) -> WmsgB[k][e*128+d] (bf16)
__global__ void prep_wmsg(const float* __restrict__ wsrc, bf16* __restrict__ o)
{
    int idx = blockIdx.x * 256 + threadIdx.x;
    int e = idx >> 14, rem = idx & 16383;
    int k = rem >> 7, d = rem & 127;
    o[(size_t)k * (NE * DIM) + e * DIM + d] = (bf16)wsrc[idx];
}

__global__ void f32_to_bf16(const float* __restrict__ in, bf16* __restrict__ out, int n)
{
    int i = blockIdx.x * 256 + threadIdx.x;
    if (i < n) out[i] = (bf16)in[i];
}

// GRU gating: h_new = (1-z)*tanh(i_n + r*h_n) + z*h
__global__ void __launch_bounds__(256)
gru_gate(const float* __restrict__ gi, const float* __restrict__ gh,
         const float* __restrict__ h,
         float* __restrict__ h_out, bf16* __restrict__ hb_out)
{
    int idx = blockIdx.x * 256 + threadIdx.x;
    int row = idx >> 7, d = idx & 127;
    const float* gir = gi + (size_t)row * (3 * DIM);
    const float* ghr = gh + (size_t)row * (3 * DIM);
    float ir = gir[d], iz = gir[DIM + d], in_ = gir[2 * DIM + d];
    float hr = ghr[d], hz = ghr[DIM + d], hn  = ghr[2 * DIM + d];
    float r = 1.f / (1.f + __expf(-(ir + hr)));
    float z = 1.f / (1.f + __expf(-(iz + hz)));
    float n = tanhf(in_ + r * hn);
    float o = (1.f - z) * n + z * h[idx];
    h_out[idx]  = o;
    hb_out[idx] = (bf16)o;
}

// ---------------------------------------------------------------------------
extern "C" void kernel_launch(void* const* d_in, const int* in_sizes, int n_in,
                              void* d_out, int out_size, void* d_ws, size_t ws_size,
                              hipStream_t stream)
{
    const float* x_node = (const float*)d_in[0];
    const float* x_edge = (const float*)d_in[1];
    const float* x_wmsg = (const float*)d_in[2];
    const float* x_bmsg = (const float*)d_in[3];
    const float* x_wi   = (const float*)d_in[4];
    const float* x_wh   = (const float*)d_in[5];
    const float* x_bgru = (const float*)d_in[6];
    const float* x_ar   = (const float*)d_in[7];
    float* out = (float*)d_out;

    char* ws = (char*)d_ws;
    size_t off = 0;
    auto carve = [&](size_t bytes) -> char* {
        char* p = ws + off;
        off = (off + bytes + 255) & ~(size_t)255;
        return p;
    };

    bf16*  edgeT = (bf16*)carve((size_t)NND * KAGG * 2);             // 64 MiB
    float* parts = (float*)carve((size_t)NSPLIT * NND * DIM * 4);    // 16 MiB
    bf16*  Pm    = (bf16*)carve((size_t)NND * NE * DIM * 2);
    bf16*  WmsgB = (bf16*)carve((size_t)DIM * NE * DIM * 2);
    bf16*  WiB   = (bf16*)carve((size_t)DIM * 3 * DIM * 2);
    bf16*  WhB   = (bf16*)carve((size_t)DIM * 3 * DIM * 2);
    bf16*  ArB   = (bf16*)carve((size_t)DIM * DIM * 2);
    float* h0    = (float*)carve((size_t)NND * DIM * 4);
    float* h1    = (float*)carve((size_t)NND * DIM * 4);
    bf16*  hb    = (bf16*)carve((size_t)NND * DIM * 2);
    bf16*  msgb  = (bf16*)carve((size_t)NND * DIM * 2);
    float* gi    = (float*)carve((size_t)NND * 3 * DIM * 4);
    float* gh    = (float*)carve((size_t)NND * 3 * DIM * 4);
    bf16*  Tb    = (bf16*)carve((size_t)NND * DIM * 2);
    (void)ws_size; (void)in_sizes; (void)n_in; (void)out_size;

    // ---- one-time prep ----
    hipMemcpyAsync(h0, x_node, (size_t)NND * DIM * 4, hipMemcpyDeviceToDevice, stream);
    f32_to_bf16<<<(NND * DIM) / 256, 256, 0, stream>>>(x_node, hb, NND * DIM);
    prep_wmsg<<<(NE * DIM * DIM) / 256, 256, 0, stream>>>(x_wmsg, WmsgB);
    f32_to_bf16<<<(DIM * 3 * DIM) / 256, 256, 0, stream>>>(x_wi, WiB, DIM * 3 * DIM);
    f32_to_bf16<<<(DIM * 3 * DIM) / 256, 256, 0, stream>>>(x_wh, WhB, DIM * 3 * DIM);
    f32_to_bf16<<<(DIM * DIM) / 256, 256, 0, stream>>>(x_ar, ArB, DIM * DIM);
    edge_transpose<<<dim3(NND / 16, NND / 16), 256, 0, stream>>>(x_edge, edgeT);

    float* hcur = h0;
    float* hnxt = h1;
    for (int it = 0; it < NITER; ++it) {
        // P = h @ WmsgB : [2048 x 1024], K=128
        wmma_gemm<false, false, false, true><<<dim3(32, NND / 128), 256, 0, stream>>>(
            hb, DIM, WmsgB, NE * DIM, nullptr, Pm, NE * DIM, DIM, NND);
        // partials[z] = edgeT @ P over K-slice z : split-K over K=16384
        wmma_gemm<false, true, false, false><<<dim3(DIM / 32, NND / 128, NSPLIT), 256, 0, stream>>>(
            edgeT, KAGG, Pm, DIM, nullptr, parts, DIM, KAGG / NSPLIT, NND);
        // msg = sum(partials) + b_msg -> bf16
        reduce_msg<<<(NND * DIM) / 256, 256, 0, stream>>>(parts, x_bmsg, msgb);
        // gi = msg @ Wi + b_gru : [2048 x 384], K=128
        wmma_gemm<false, false, true, false><<<dim3(3 * DIM / 32, NND / 128), 256, 0, stream>>>(
            msgb, DIM, WiB, 3 * DIM, x_bgru, gi, 3 * DIM, DIM, NND);
        // gh = h @ Wh : [2048 x 384], K=128
        wmma_gemm<false, false, false, false><<<dim3(3 * DIM / 32, NND / 128), 256, 0, stream>>>(
            hb, DIM, WhB, 3 * DIM, nullptr, gh, 3 * DIM, DIM, NND);
        gru_gate<<<(NND * DIM) / 256, 256, 0, stream>>>(gi, gh, hcur, hnxt, hb);
        float* t = hcur; hcur = hnxt; hnxt = t;
    }

    // T = h @ A_readout : [2048 x 128], K=128
    wmma_gemm<false, false, false, true><<<dim3(DIM / 32, NND / 128), 256, 0, stream>>>(
        hb, DIM, ArB, DIM, nullptr, Tb, DIM, DIM, NND);
    // logits = T @ h^T : [2048 x 2048], K=128
    wmma_gemm<true, false, false, false><<<dim3(NND / 32, NND / 128), 256, 0, stream>>>(
        Tb, DIM, hb, DIM, nullptr, out, NND, DIM, NND);
}
